// YoloSimpleOTATarget_27273042329993
// MI455X (gfx1250) — matrix operand order repaired and checked by reading, not compiled, not run on hardware
//
#include <hip/hip_runtime.h>
#include <hip/hip_bf16.h>
#include <math.h>

// ---------------- problem constants ----------------
#define B_    16
#define G_    32
#define M_    320
#define N_    960          // 3*M_
#define NC    80
#define CH_   85
#define KPAD  96           // 80 classes padded to 3 WMMA K-chunks of 32
#define TOPK  20

typedef __attribute__((ext_vector_type(16))) _Float16 v16h;
typedef __attribute__((ext_vector_type(8)))  float    v8f;

// ---------------- workspace layout (bytes) ----------------
// zbuf : _Float16 [B][N][KPAD]          = 2,949,120 B
// S    : float    [B][N]                =    61,440 B
// pbox : float    [B][N][4]             =   245,760 B
// gbox : float    [B][G][4]             =     8,192 B
// gcls : int      [B][G]                =     2,048 B
// dk   : int      [B][G]                =     2,048 B
static constexpr size_t WS_Z    = 0;
static constexpr size_t WS_S    = WS_Z    + (size_t)B_ * N_ * KPAD * 2;
static constexpr size_t WS_PBOX = WS_S    + (size_t)B_ * N_ * 4;
static constexpr size_t WS_GBOX = WS_PBOX + (size_t)B_ * N_ * 4 * 4;
static constexpr size_t WS_GCLS = WS_GBOX + (size_t)B_ * G_ * 4 * 4;
static constexpr size_t WS_DK   = WS_GCLS + (size_t)B_ * G_ * 4;

// ---------------- output layout (floats) ----------------
static constexpr size_t OUT_MATCH = 0;                       // (B,G,N)
static constexpr size_t OUT_FG    = (size_t)B_ * G_ * N_;    // (B,N)
static constexpr size_t OUT_MG    = OUT_FG + (size_t)B_ * N_;// (B,N)
static constexpr size_t OUT_COST  = OUT_MG + (size_t)B_ * N_;// (B,G,N)

__device__ __forceinline__ float iou_box(const float* a, const float* b) {
    float lx = fmaxf(a[0], b[0]), ly = fmaxf(a[1], b[1]);
    float rx = fminf(a[2], b[2]), ry = fminf(a[3], b[3]);
    float w  = fmaxf(rx - lx, 0.f), h = fmaxf(ry - ly, 0.f);
    float inter = w * h;
    float aa = (a[2] - a[0]) * (a[3] - a[1]);
    float ab = (b[2] - b[0]) * (b[3] - b[1]);
    return inter / (aa + ab - inter);
}

// ---------------------------------------------------------------------------
// Kernel 1: gather candidates; compute pred xyxy boxes, S(b,n), z(b,n,c)->f16
// ---------------------------------------------------------------------------
__global__ void prep_cand(const float* __restrict__ pred0,
                          const float* __restrict__ pred1,
                          const float* __restrict__ pred2,
                          const int* __restrict__ a0, const int* __restrict__ a1,
                          const int* __restrict__ a2,
                          const int* __restrict__ gj0, const int* __restrict__ gj1,
                          const int* __restrict__ gj2,
                          const int* __restrict__ gi0, const int* __restrict__ gi1,
                          const int* __restrict__ gi2,
                          _Float16* __restrict__ zbuf,
                          float* __restrict__ S,
                          float* __restrict__ pbox) {
    int t = blockIdx.x * blockDim.x + threadIdx.x;
    if (t >= B_ * N_) return;
    int b = t / N_, n = t % N_;
    int l = n / M_, m = n % M_;

    const float* pred; const int *aa, *jj, *ii; int hdim; float stride;
    if (l == 0)      { pred = pred0; aa = a0; jj = gj0; ii = gi0; hdim = 80; stride = 8.f;  }
    else if (l == 1) { pred = pred1; aa = a1; jj = gj1; ii = gi1; hdim = 40; stride = 16.f; }
    else             { pred = pred2; aa = a2; jj = gj2; ii = gi2; hdim = 20; stride = 32.f; }

    int a  = aa[b * M_ + m];
    int gj = jj[b * M_ + m];
    int gi = ii[b * M_ + m];
    const float* c = pred + ((((size_t)b * 3 + a) * hdim + gj) * hdim + gi) * CH_;

    float cx = c[0] * stride, cy = c[1] * stride;
    float w  = c[2] * stride, hh = c[3] * stride;
    float* pb = pbox + (size_t)t * 4;
    pb[0] = cx - 0.5f * w;  pb[1] = cy - 0.5f * hh;
    pb[2] = cx + 0.5f * w;  pb[3] = cy + 0.5f * hh;

    float obj = c[4];
    _Float16* zr = zbuf + (size_t)t * KPAD;
    float s = 0.f;
    for (int k = 0; k < NC; ++k) {
        float y = sqrtf(c[5 + k] * obj);
        float z = logf(y / (1.f - y));
        s += fmaxf(z, 0.f) + log1pf(expf(-fabsf(z)));
        zr[k] = (_Float16)z;
    }
    for (int k = NC; k < KPAD; ++k) zr[k] = (_Float16)0.f;  // K padding
    S[t] = s;
}

// ---------------------------------------------------------------------------
// Kernel 2: GT boxes (xyxy * 640) and class ids
// ---------------------------------------------------------------------------
__global__ void prep_gt(const float* __restrict__ targets,
                        float* __restrict__ gbox, int* __restrict__ gcls) {
    int t = blockIdx.x * blockDim.x + threadIdx.x;
    if (t >= B_ * G_) return;
    const float* tg = targets + (size_t)t * 6;
    gcls[t] = (int)tg[1];
    float cx = tg[2] * 640.f, cy = tg[3] * 640.f;
    float w  = tg[4] * 640.f, h  = tg[5] * 640.f;
    float* gb = gbox + (size_t)t * 4;
    gb[0] = cx - 0.5f * w;  gb[1] = cy - 0.5f * h;
    gb[2] = cx + 0.5f * w;  gb[3] = cy + 0.5f * h;
}

// ---------------------------------------------------------------------------
// Kernel 3: dynamic-k = max(int(sum of top-20 IoU), 1) per (b,g)
// ---------------------------------------------------------------------------
__global__ void compute_dk(const float* __restrict__ gbox,
                           const float* __restrict__ pbox,
                           int* __restrict__ dk) {
    int t = blockIdx.x * blockDim.x + threadIdx.x;
    if (t >= B_ * G_) return;
    int b = t / G_;
    float gb[4];
    gb[0] = gbox[t * 4 + 0]; gb[1] = gbox[t * 4 + 1];
    gb[2] = gbox[t * 4 + 2]; gb[3] = gbox[t * 4 + 3];

    float top[TOPK];
#pragma unroll
    for (int k = 0; k < TOPK; ++k) top[k] = -1.0e30f;

    for (int n = 0; n < N_; ++n) {
        const float* pb = pbox + ((size_t)b * N_ + n) * 4;
        float v = iou_box(gb, pb);
        if (v > top[TOPK - 1]) {          // descending insertion (keep 20 largest)
            int p = TOPK - 1;
            while (p > 0 && top[p - 1] < v) { top[p] = top[p - 1]; --p; }
            top[p] = v;
        }
    }
    float s = 0.f;
#pragma unroll
    for (int k = 0; k < TOPK; ++k) s += top[k];
    int d = (int)s;                       // trunc toward zero (s >= 0)
    if (d < 1) d = 1;
    dk[t] = d;
}

// ---------------------------------------------------------------------------
// Kernel 4: WMMA cost tiles.  One wave32 per (b, 16-row GT tile, 16-col cand
// tile).  D = T_onehot(16x96,f16) x Z^T(96x16,f16) via 3x v_wmma_f32_16x16x32_f16,
// then fuse S(b,n), -3*log(iou+1e-5) and write cost to d_out.
// ---------------------------------------------------------------------------
__global__ void __launch_bounds__(32) wmma_cost(
        const _Float16* __restrict__ zbuf,
        const float* __restrict__ S,
        const float* __restrict__ pbox,
        const float* __restrict__ gbox,
        const int* __restrict__ gcls,
        float* __restrict__ cost) {
    const int b    = blockIdx.x;
    const int mt   = blockIdx.y;   // GT tile  (0..1)
    const int nt   = blockIdx.z;   // cand tile (0..59)
    const int lane = threadIdx.x;
    const int mlo  = lane & 15;
    const bool hi  = lane >= 16;

    const int n   = nt * 16 + mlo;                 // candidate column
    const int ga  = mt * 16 + mlo;                 // A-matrix row for this lane
    const int cls = gcls[b * G_ + ga];
    const _Float16* zr = zbuf + ((size_t)b * N_ + n) * KPAD;

    __builtin_prefetch(zr, 0, 1);                  // global_prefetch_b8 of z row

    v8f acc = {};
#pragma unroll
    for (int kc = 0; kc < 3; ++kc) {
        const int kbase = kc * 32;
        v16h A{}; v16h Bm{};
#pragma unroll
        for (int j = 0; j < 8; ++j) {
            // A (16x32 f16): VGPR j holds K = 2j(+8 for j>=4); hi lanes K+8.
            int kl = 2 * j + (j >= 4 ? 8 : 0);
            int ka = kbase + kl + (hi ? 8 : 0);
            A[2 * j]     = (cls == ka)     ? (_Float16)1.f : (_Float16)0.f;
            A[2 * j + 1] = (cls == ka + 1) ? (_Float16)1.f : (_Float16)0.f;
            // B (32x16 f16): VGPR j holds K = 2j(+8 for j>=4); hi lanes K+8.
            int kb = kbase + 2 * j + (j >= 4 ? 8 : 0) + (hi ? 8 : 0);
            Bm[2 * j]     = zr[kb];
            Bm[2 * j + 1] = zr[kb + 1];
        }
        acc = __builtin_amdgcn_wmma_f32_16x16x32_f16(
            /*neg_a=*/false, A, /*neg_b=*/false, Bm,
            /*c_mod=*/(short)0, acc, /*reuse_a=*/false, /*reuse_b=*/false);
    }

    // fuse: cost(b,g,n) = S(b,n) - dot + 3 * (-log(iou + 1e-5))
    float pb[4];
    const float* pbp = pbox + ((size_t)b * N_ + n) * 4;
    pb[0] = pbp[0]; pb[1] = pbp[1]; pb[2] = pbp[2]; pb[3] = pbp[3];
    const float sn = S[b * N_ + n];

#pragma unroll
    for (int r = 0; r < 8; ++r) {
        int g = mt * 16 + r + (hi ? 8 : 0);        // C/D layout: hi lanes M=r+8
        const float* gb = gbox + ((size_t)b * G_ + g) * 4;
        float iou = iou_box(gb, pb);
        float v = (sn - acc[r]) + 3.f * (-logf(iou + 1e-5f));
        cost[((size_t)b * G_ + g) * N_ + n] = v;
    }
}

// ---------------------------------------------------------------------------
// Kernel 5: per-(b,g) row: top-20 smallest costs, set matching[pos[k<dk]] = 1
// ---------------------------------------------------------------------------
__global__ void match_rows(const float* __restrict__ cost,
                           const int* __restrict__ dk,
                           float* __restrict__ matching) {
    int t = blockIdx.x * blockDim.x + threadIdx.x;
    if (t >= B_ * G_) return;
    const float* cr = cost + (size_t)t * N_;
    float* mr = matching + (size_t)t * N_;

    float val[TOPK]; int idx[TOPK];
#pragma unroll
    for (int k = 0; k < TOPK; ++k) { val[k] = 1.0e30f; idx[k] = 0; }

    for (int n = 0; n < N_; ++n) {
        mr[n] = 0.f;
        float v = cr[n];
        if (v < val[TOPK - 1]) {          // ascending insertion, first-index wins ties
            int p = TOPK - 1;
            while (p > 0 && val[p - 1] > v) {
                val[p] = val[p - 1]; idx[p] = idx[p - 1]; --p;
            }
            val[p] = v; idx[p] = n;
        }
    }
    int d = dk[t]; if (d > TOPK) d = TOPK;
    for (int k = 0; k < d; ++k) mr[idx[k]] = 1.f;
}

// ---------------------------------------------------------------------------
// Kernel 6: per-(b,n) column: conflict resolution, fg_mask, matched_gt
// ---------------------------------------------------------------------------
__global__ void resolve_cols(const float* __restrict__ cost,
                             float* __restrict__ matching,
                             float* __restrict__ fg,
                             float* __restrict__ mg) {
    int t = blockIdx.x * blockDim.x + threadIdx.x;
    if (t >= B_ * N_) return;
    int b = t / N_, n = t % N_;

    float cnt = 0.f; int firstg = -1;
    for (int g = 0; g < G_; ++g) {
        float v = matching[((size_t)b * G_ + g) * N_ + n];
        cnt += v;
        if (firstg < 0 && v > 0.f) firstg = g;
    }
    if (cnt > 1.f) {
        float best = cost[((size_t)b * G_) * N_ + n]; int bg = 0;
        for (int g = 1; g < G_; ++g) {
            float v = cost[((size_t)b * G_ + g) * N_ + n];
            if (v < best) { best = v; bg = g; }
        }
        for (int g = 0; g < G_; ++g)
            matching[((size_t)b * G_ + g) * N_ + n] = (g == bg) ? 1.f : 0.f;
        fg[t] = 1.f;
        mg[t] = (float)bg;
    } else {
        fg[t] = (cnt > 0.f) ? 1.f : 0.f;
        mg[t] = (firstg >= 0) ? (float)firstg : 0.f;
    }
}

// ---------------------------------------------------------------------------
extern "C" void kernel_launch(void* const* d_in, const int* in_sizes, int n_in,
                              void* d_out, int out_size, void* d_ws, size_t ws_size,
                              hipStream_t stream) {
    (void)in_sizes; (void)n_in; (void)out_size; (void)ws_size;

    // setup_inputs() dict order:
    // pred0,a0,gj0,gi0,anc0, pred1,a1,gj1,gi1,anc1, pred2,a2,gj2,gi2,anc2, targets
    const float* pred0 = (const float*)d_in[0];
    const int*   a0    = (const int*)  d_in[1];
    const int*   gj0   = (const int*)  d_in[2];
    const int*   gi0   = (const int*)  d_in[3];
    const float* pred1 = (const float*)d_in[5];
    const int*   a1    = (const int*)  d_in[6];
    const int*   gj1   = (const int*)  d_in[7];
    const int*   gi1   = (const int*)  d_in[8];
    const float* pred2 = (const float*)d_in[10];
    const int*   a2    = (const int*)  d_in[11];
    const int*   gj2   = (const int*)  d_in[12];
    const int*   gi2   = (const int*)  d_in[13];
    const float* targets = (const float*)d_in[15];
    // anc0/1/2 (d_in[4],[9],[14]) are unused by the reference.

    char* ws = (char*)d_ws;
    _Float16* zbuf = (_Float16*)(ws + WS_Z);
    float*    Sb   = (float*)   (ws + WS_S);
    float*    pbox = (float*)   (ws + WS_PBOX);
    float*    gbox = (float*)   (ws + WS_GBOX);
    int*      gcls = (int*)     (ws + WS_GCLS);
    int*      dk   = (int*)     (ws + WS_DK);

    float* out      = (float*)d_out;
    float* matching = out + OUT_MATCH;
    float* fg       = out + OUT_FG;
    float* mg       = out + OUT_MG;
    float* cost     = out + OUT_COST;

    prep_cand<<<(B_ * N_ + 255) / 256, 256, 0, stream>>>(
        pred0, pred1, pred2, a0, a1, a2, gj0, gj1, gj2, gi0, gi1, gi2,
        zbuf, Sb, pbox);

    prep_gt<<<(B_ * G_ + 255) / 256, 256, 0, stream>>>(targets, gbox, gcls);

    compute_dk<<<(B_ * G_ + 255) / 256, 256, 0, stream>>>(gbox, pbox, dk);

    wmma_cost<<<dim3(B_, 2, N_ / 16), 32, 0, stream>>>(
        zbuf, Sb, pbox, gbox, gcls, cost);

    match_rows<<<(B_ * G_ + 255) / 256, 256, 0, stream>>>(cost, dk, matching);

    resolve_cols<<<(B_ * N_ + 255) / 256, 256, 0, stream>>>(cost, matching, fg, mg);
}
